// DConvModules_31559419691502
// MI455X (gfx1250) — matrix operand order repaired and checked by reading, not compiled, not run on hardware
//
#include <hip/hip_runtime.h>
#include <hip/hip_bf16.h>
#include <math.h>

typedef __bf16 v16bf __attribute__((ext_vector_type(16)));
typedef __bf16 v8bf  __attribute__((ext_vector_type(8)));
typedef float  v8f   __attribute__((ext_vector_type(8)));

#define B_    4
#define CE_   32
#define CF_   64
#define CIN_  96
#define H_    128
#define W_    128
#define HW_   (H_*W_)
#define KK_   9
#define KSTEP_ 27          /* 864 / 32 ; ks = kk*3 + j, cbase = j*32 */
#define NT_   32           /* pixels per workgroup (two 16-wide WMMA tiles) */
#define EPS   1e-5f
#define LRELU 0.2f

__device__ __forceinline__ unsigned short f2bf(float f) {
  unsigned int u = __float_as_uint(f);
  u = (u + 0x7FFFu + ((u >> 16) & 1u)) >> 16;   // round-to-nearest-even
  return (unsigned short)u;
}
__device__ __forceinline__ __bf16 f2bf16(float f) {
  return __builtin_bit_cast(__bf16, f2bf(f));
}
__device__ __forceinline__ float bf2f(unsigned short u) {
  return __uint_as_float((unsigned)u << 16);
}
__device__ __forceinline__ int clampi(int v, int lo, int hi) {
  return v < lo ? lo : (v > hi ? hi : v);
}

// K-permutation shared by packers and column builders: ck = kk*96 + c.
// K-step ks in [0,27): kk = ks/3 (scalar), c = (ks%3)*32 + klocal.
// A-fragment (16-bit A 16x32, ISA 7.12.2):
//   row M = lane&15 ; kb = (lane>>4)*8 ; e<8 -> K=kb+e ; e>=8 -> K=16+kb+e-8

__global__ void pack_reg_kernel(const float* __restrict__ w,
                                unsigned short* __restrict__ out) {
  int t = blockIdx.x * blockDim.x + threadIdx.x;
  if (t >= 4 * KSTEP_ * 32 * 16) return;
  int e    = t & 15;
  int lane = (t >> 4) & 31;
  int rest = t >> 9;
  int ks   = rest % KSTEP_;
  int mt   = rest / KSTEP_;
  int kb   = (lane >> 4) * 8;
  int klocal = (e < 8) ? (kb + e) : (16 + kb + (e - 8));
  int kk = ks / 3;
  int c  = (ks - 3 * kk) * 32 + klocal;
  int o  = mt * 16 + (lane & 15);
  out[t] = f2bf(w[((size_t)o * CIN_ + c) * KK_ + kk]);
}

__global__ void pack_om_kernel(const float* __restrict__ off_w,
                               const float* __restrict__ mod_w,
                               unsigned short* __restrict__ out) {
  int t = blockIdx.x * blockDim.x + threadIdx.x;
  if (t >= 2 * KSTEP_ * 32 * 16) return;
  int e    = t & 15;
  int lane = (t >> 4) & 31;
  int rest = t >> 9;
  int ks   = rest % KSTEP_;
  int mt   = rest / KSTEP_;
  int kb   = (lane >> 4) * 8;
  int klocal = (e < 8) ? (kb + e) : (16 + kb + (e - 8));
  int kk = ks / 3;
  int c  = (ks - 3 * kk) * 32 + klocal;
  int o  = mt * 16 + (lane & 15);
  float v = 0.f;
  if (o < 18)      v = off_w[((size_t)o * CIN_ + c) * KK_ + kk];
  else if (o < 27) v = mod_w[((size_t)(o - 18) * CIN_ + c) * KK_ + kk];
  out[t] = f2bf(v);
}

// ---------------------------------------------------------------------------
// Channel-LN of both inputs, concatenated into ox (B,96,H,W) **bf16**
// (the GEMMs consume bf16 anyway; halves the L2 gather traffic downstream)
// ---------------------------------------------------------------------------
__global__ void ln_concat_kernel(const float* __restrict__ evt,
                                 const float* __restrict__ ef,
                                 const float* __restrict__ wE, const float* __restrict__ bE,
                                 const float* __restrict__ wF, const float* __restrict__ bF,
                                 unsigned short* __restrict__ ox) {
  int p = blockIdx.x * blockDim.x + threadIdx.x;
  if (p >= B_ * HW_) return;
  int b = p / HW_, pix = p - b * HW_;

  const float* xe = evt + (size_t)b * CE_ * HW_ + pix;
  float s = 0.f, sq = 0.f;
  for (int c = 0; c < CE_; ++c) { float v = xe[(size_t)c * HW_]; s += v; sq += v * v; }
  float mu  = s / CE_;
  float inv = rsqrtf(sq / CE_ - mu * mu + EPS);
  unsigned short* oxe = ox + (size_t)b * CIN_ * HW_ + pix;
  for (int c = 0; c < CE_; ++c)
    oxe[(size_t)c * HW_] = f2bf((xe[(size_t)c * HW_] - mu) * inv * wE[c] + bE[c]);

  const float* xf = ef + (size_t)b * CF_ * HW_ + pix;
  s = 0.f; sq = 0.f;
  for (int c = 0; c < CF_; ++c) { float v = xf[(size_t)c * HW_]; s += v; sq += v * v; }
  mu  = s / CF_;
  inv = rsqrtf(sq / CF_ - mu * mu + EPS);
  unsigned short* oxf = oxe + (size_t)CE_ * HW_;
  for (int c = 0; c < CF_; ++c)
    oxf[(size_t)c * HW_] = f2bf((xf[(size_t)c * HW_] - mu) * inv * wF[c] + bF[c]);
}

// ---------------------------------------------------------------------------
// 3x3 conv (offset 18ch + mask 9ch): im2col GEMM, M=32(27), N=32, K=864.
// 2 waves. Column-build: thread owns pixel n = t&31 and K-rows
// klocal = (t>>5)*8 + e (e<8), 16 + (t>>5)*8 + e-8 (e>=8)  -> x-clamp/validity computed once per (kk,thread);
// 8-element runs stored as one 16B ds_store.
// ---------------------------------------------------------------------------
__global__ void __launch_bounds__(64)
offmask_gemm_kernel(const unsigned short* __restrict__ ox,
                    const unsigned short* __restrict__ packOM,
                    const float* __restrict__ off_b, const float* __restrict__ mod_b,
                    float* __restrict__ offs, float* __restrict__ maskb) {
  __shared__ __attribute__((aligned(32))) __bf16 ldsB[2 * 512];
  const int t = threadIdx.x;
  const int lane = t & 31, wave = t >> 5;
  const int b = blockIdx.y;
  const int base = blockIdx.x * NT_;
  const int y = base / W_, x0 = base % W_;
  const unsigned short* oxb = ox + (size_t)b * CIN_ * HW_;

  const int n    = t & 31;          // pixel column owned by this thread
  const int q    = t >> 5;          // 0..1 -> K-half
  const int lu   = q * 16 + (n & 15);
  const int half = n >> 4;
  __bf16* wr = &ldsB[half * 512 + lu * 16];

  const v16bf* pA = reinterpret_cast<const v16bf*>(packOM)
                    + (size_t)wave * KSTEP_ * 32 + lane;
  v8f acc0 = {}, acc1 = {};
  for (int kk = 0; kk < KK_; ++kk) {
    const int ky = kk / 3 - 1, kx = kk - 3 * (kk / 3) - 1;
    const int yy = y + ky;
    const float rowv = (yy >= 0 && yy < H_) ? 1.f : 0.f;
    const unsigned short* rowp = oxb + (size_t)clampi(yy, 0, H_ - 1) * W_;
    const int   xx  = x0 + n + kx;
    const float xv  = (xx >= 0 && xx < W_) ? rowv : 0.f;
    const int   xxc = clampi(xx, 0, W_ - 1);
#pragma unroll
    for (int j = 0; j < 3; ++j) {
      const int cthr = j * 32 + q * 8;    // base channel for this thread
#pragma unroll
      for (int h = 0; h < 2; ++h) {
        v8bf tmp;
#pragma unroll
        for (int ee = 0; ee < 8; ++ee) {
          int c = cthr + h * 16 + ee;
          tmp[ee] = f2bf16(bf2f(rowp[(size_t)c * HW_ + xxc]) * xv);
        }
        *reinterpret_cast<v8bf*>(wr + h * 8) = tmp;
      }
      v16bf a = pA[(size_t)(kk * 3 + j) * 32];
      __builtin_prefetch(pA + (size_t)(kk * 3 + j + 1) * 32, 0, 1);
      __syncthreads();
      v16bf b0 = *reinterpret_cast<const v16bf*>(&ldsB[lane * 16]);
      v16bf b1 = *reinterpret_cast<const v16bf*>(&ldsB[512 + lane * 16]);
      acc0 = __builtin_amdgcn_wmma_f32_16x16x32_bf16(false, a, false, b0,
                                                     (short)0, acc0, false, false);
      acc1 = __builtin_amdgcn_wmma_f32_16x16x32_bf16(false, a, false, b1,
                                                     (short)0, acc1, false, false);
      __syncthreads();
    }
  }
  for (int r = 0; r < 8; ++r) {
    int M = r + ((lane >> 4) << 3);
    int o = wave * 16 + M;
    int pix = base + (lane & 15);
    if (o < 18) {
      float bb = off_b[o];
      size_t p0 = ((size_t)b * 18 + o) * HW_ + pix;
      offs[p0]      = acc0[r] + bb;
      offs[p0 + 16] = acc1[r] + bb;
    } else if (o < 27) {
      int qq = o - 18;
      float bb = mod_b[qq];
      size_t p0 = ((size_t)b * KK_ + qq) * HW_ + pix;
      maskb[p0]      = 2.f / (1.f + __expf(-(acc0[r] + bb)));
      maskb[p0 + 16] = 2.f / (1.f + __expf(-(acc1[r] + bb)));
    }
  }
}

// ---------------------------------------------------------------------------
// Deformable conv GEMM: M=64 (4 waves x 16), N=32, K=864.
// Thread owns pixel n = t&31 and an 8-wide K run; the 4 bilinear gather
// indices + weights are loaded from LDS once per (kk,thread) and reused
// across all 24 channel elements of that kk.
// ---------------------------------------------------------------------------
__global__ void __launch_bounds__(128)
deform_gemm_kernel(const unsigned short* __restrict__ ox,
                   const unsigned short* __restrict__ packReg,
                   const float* __restrict__ offs, const float* __restrict__ maskb,
                   const float* __restrict__ reg_b, float* __restrict__ zbuf) {
  __shared__ __attribute__((aligned(32))) __bf16 ldsB[2 * 512];
  __shared__ int   sIdx[4][KK_ * NT_];
  __shared__ float sWgt[4][KK_ * NT_];
  const int t = threadIdx.x;
  const int lane = t & 31, wave = t >> 5;
  const int b = blockIdx.y;
  const int base = blockIdx.x * NT_;
  const int y = base / W_, x0 = base % W_;
  const unsigned short* oxb = ox + (size_t)b * CIN_ * HW_;

  for (int i = t; i < KK_ * NT_; i += 128) {
    int kk = i >> 5, n = i & (NT_ - 1);
    int pix = base + n;
    float dy = offs[((size_t)b * 18 + kk * 2 + 0) * HW_ + pix];
    float dx = offs[((size_t)b * 18 + kk * 2 + 1) * HW_ + pix];
    float m  = maskb[((size_t)b * KK_ + kk) * HW_ + pix];
    float py = (float)y + (float)(kk / 3 - 1) + dy;
    float px = (float)(x0 + n) + (float)(kk % 3 - 1) + dx;
    float fy = floorf(py), fx = floorf(px);
    float wy1 = py - fy, wx1 = px - fx;
    float cw[4] = { (1.f - wy1) * (1.f - wx1), (1.f - wy1) * wx1,
                    wy1 * (1.f - wx1),          wy1 * wx1 };
    float cy[4] = { fy, fy, fy + 1.f, fy + 1.f };
    float cx[4] = { fx, fx + 1.f, fx, fx + 1.f };
#pragma unroll
    for (int jj = 0; jj < 4; ++jj) {
      bool valid = (cy[jj] >= 0.f) && (cy[jj] <= (float)(H_ - 1)) &&
                   (cx[jj] >= 0.f) && (cx[jj] <= (float)(W_ - 1));
      int yi = (int)fminf(fmaxf(cy[jj], 0.f), (float)(H_ - 1));
      int xi = (int)fminf(fmaxf(cx[jj], 0.f), (float)(W_ - 1));
      sIdx[jj][i] = yi * W_ + xi;
      sWgt[jj][i] = valid ? cw[jj] * m : 0.f;
    }
  }
  __syncthreads();

  const int n    = t & 31;          // pixel column owned by this thread
  const int q    = t >> 5;          // 0..3: K-half (q&1), e-half (q>>1)
  const int lu   = (q & 1) * 16 + (n & 15);
  const int half = n >> 4;
  const int e0   = (q >> 1) * 8;
  const int klb  = (q >> 1) * 16 + (q & 1) * 8;   // klocal base of this thread's run
  __bf16* wr = &ldsB[half * 512 + lu * 16 + e0];

  const v16bf* pA = reinterpret_cast<const v16bf*>(packReg)
                    + (size_t)wave * KSTEP_ * 32 + lane;
  v8f acc0 = {}, acc1 = {};
  for (int kk = 0; kk < KK_; ++kk) {
    const int mi = kk * NT_ + n;
    const int   i0 = sIdx[0][mi], i1 = sIdx[1][mi], i2 = sIdx[2][mi], i3 = sIdx[3][mi];
    const float w0 = sWgt[0][mi], w1 = sWgt[1][mi], w2 = sWgt[2][mi], w3 = sWgt[3][mi];
#pragma unroll
    for (int j = 0; j < 3; ++j) {
      const int cthr = j * 32 + klb;
      v8bf tmp;
#pragma unroll
      for (int ee = 0; ee < 8; ++ee) {
        const unsigned short* oc = oxb + (size_t)(cthr + ee) * HW_;
        float v = bf2f(oc[i0]) * w0 + bf2f(oc[i1]) * w1
                + bf2f(oc[i2]) * w2 + bf2f(oc[i3]) * w3;
        tmp[ee] = f2bf16(v);
      }
      *reinterpret_cast<v8bf*>(wr) = tmp;
      v16bf a = pA[(size_t)(kk * 3 + j) * 32];
      __builtin_prefetch(pA + (size_t)(kk * 3 + j + 1) * 32, 0, 1);
      __syncthreads();
      v16bf b0 = *reinterpret_cast<const v16bf*>(&ldsB[lane * 16]);
      v16bf b1 = *reinterpret_cast<const v16bf*>(&ldsB[512 + lane * 16]);
      acc0 = __builtin_amdgcn_wmma_f32_16x16x32_bf16(false, a, false, b0,
                                                     (short)0, acc0, false, false);
      acc1 = __builtin_amdgcn_wmma_f32_16x16x32_bf16(false, a, false, b1,
                                                     (short)0, acc1, false, false);
      __syncthreads();
    }
  }
  for (int r = 0; r < 8; ++r) {
    int M = r + ((lane >> 4) << 3);
    int o = wave * 16 + M;
    int pix = base + (lane & 15);
    float bb = reg_b[o];
    float v0 = acc0[r] + bb;
    float v1 = acc1[r] + bb;
    v0 = (v0 >= 0.f) ? v0 : LRELU * v0;
    v1 = (v1 >= 0.f) ? v1 : LRELU * v1;
    size_t p0 = ((size_t)b * CF_ + o) * HW_ + pix;
    zbuf[p0]      = v0;
    zbuf[p0 + 16] = v1;
  }
}

// Final channel-LN over 64 channels -> d_out
__global__ void ln_out_kernel(const float* __restrict__ zbuf,
                              const float* __restrict__ w, const float* __restrict__ bias,
                              float* __restrict__ out) {
  int p = blockIdx.x * blockDim.x + threadIdx.x;
  if (p >= B_ * HW_) return;
  int b = p / HW_, pix = p - b * HW_;
  const float* z = zbuf + (size_t)b * CF_ * HW_ + pix;
  float s = 0.f, sq = 0.f;
  for (int c = 0; c < CF_; ++c) { float v = z[(size_t)c * HW_]; s += v; sq += v * v; }
  float mu  = s / CF_;
  float inv = rsqrtf(sq / CF_ - mu * mu + EPS);
  float* o = out + (size_t)b * CF_ * HW_ + pix;
  for (int c = 0; c < CF_; ++c)
    o[(size_t)c * HW_] = (z[(size_t)c * HW_] - mu) * inv * w[c] + bias[c];
}

extern "C" void kernel_launch(void* const* d_in, const int* in_sizes, int n_in,
                              void* d_out, int out_size, void* d_ws, size_t ws_size,
                              hipStream_t stream) {
  const float* f_evt    = (const float*)d_in[0];
  const float* f_ef     = (const float*)d_in[1];
  const float* ln_evt_w = (const float*)d_in[2];
  const float* ln_evt_b = (const float*)d_in[3];
  const float* ln_ef_w  = (const float*)d_in[4];
  const float* ln_ef_b  = (const float*)d_in[5];
  const float* ln_al_w  = (const float*)d_in[6];
  const float* ln_al_b  = (const float*)d_in[7];
  const float* off_w    = (const float*)d_in[8];
  const float* off_b    = (const float*)d_in[9];
  const float* mod_w    = (const float*)d_in[10];
  const float* mod_b    = (const float*)d_in[11];
  const float* reg_w    = (const float*)d_in[12];
  const float* reg_b    = (const float*)d_in[13];

  char* ws = (char*)d_ws;
  unsigned short* ox      = (unsigned short*)(ws + 0);         // 12,582,912 B
  float*          zbuf    = (float*)(ws + 12582912);           // 16,777,216 B
  float*          offs    = (float*)(ws + 29360128);           //  4,718,592 B
  float*          maskb   = (float*)(ws + 34078720);           //  2,359,296 B
  unsigned short* packReg = (unsigned short*)(ws + 36438016);  //    110,592 B
  unsigned short* packOM  = (unsigned short*)(ws + 36548608);  //     55,296 B

  pack_reg_kernel<<<216, 256, 0, stream>>>(reg_w, packReg);
  pack_om_kernel<<<108, 256, 0, stream>>>(off_w, mod_w, packOM);
  ln_concat_kernel<<<256, 256, 0, stream>>>(f_evt, f_ef, ln_evt_w, ln_evt_b,
                                            ln_ef_w, ln_ef_b, ox);
  dim3 gTiles(HW_ / NT_, B_);
  offmask_gemm_kernel<<<gTiles, 64, 0, stream>>>(ox, packOM, off_b, mod_b,
                                                 offs, maskb);
  deform_gemm_kernel<<<gTiles, 128, 0, stream>>>(ox, packReg, offs, maskb,
                                                 reg_b, zbuf);
  ln_out_kernel<<<256, 256, 0, stream>>>(zbuf, ln_al_w, ln_al_b, (float*)d_out);
}
